// DeformConvBlock_9921374454155
// MI455X (gfx1250) — matrix operand order repaired and checked by reading, not compiled, not run on hardware
//
#include <hip/hip_runtime.h>

// CDNA5 / gfx1250 deformable-conv block.
//   Kernel 1: offset conv (M=18,K=648,N=115200 GEMM) via v_wmma_f32_16x16x32_f16
//   Kernel 2: TDM offset staging -> bilinear sampling -> LDS -> grouped GEMM via WMMA

typedef __attribute__((ext_vector_type(16))) _Float16 v16h;
typedef __attribute__((ext_vector_type(8)))  float    v8f;
typedef __attribute__((ext_vector_type(4)))  unsigned int u32x4;
typedef __attribute__((ext_vector_type(8)))  int      i32x8;
typedef __attribute__((ext_vector_type(4)))  int      i32x4;

union Frag16 { v16h h; uint4 u[2]; };

// Problem constants
static constexpr int Bn  = 2;
static constexpr int Cn  = 72;
static constexpr int On  = 72;
static constexpr int Hn  = 180;
static constexpr int Wn  = 320;
static constexpr int Kt  = 9;         // 3x3 taps
static constexpr int KK1 = Cn * Kt;   // 648
static constexpr int KK1P = 672;      // padded to 21*32
static constexpr int KK2 = 8 * Kt;    // 72 (Cg*9)
static constexpr int KK2P = 96;       // padded to 3*32

// ---------------------------------------------------------------------------
// Kernel 1: offset = conv3x3(offset_feat, off_w) + off_b   -> d_ws (f32, B,18,H,W)
// One workgroup: 64 pixels of one (b,h) row. 4 waves, each a 16-pixel N-tile.
// LDS: f16 weights [32][672], f16 im2col [64][672]   (dynamic, 129 KB)
// ---------------------------------------------------------------------------
__global__ __launch_bounds__(128)
void offset_conv_wmma(const float* __restrict__ feat,
                      const float* __restrict__ off_w,
                      const float* __restrict__ off_b,
                      float* __restrict__ offs)
{
    extern __shared__ char smem1[];
    _Float16* Wlds = (_Float16*)smem1;                     // [32][KK1P]
    _Float16* Blds = (_Float16*)(smem1 + 32 * KK1P * 2);   // [64][KK1P]

    const int tid = threadIdx.x;
    const int bx  = blockIdx.x;
    const int x0  = (bx % (Wn / 64)) * 64;
    const int h   = (bx / (Wn / 64)) % Hn;
    const int b   =  bx / ((Wn / 64) * Hn);

    // zero everything with b128 stores (covers K-pad columns and M-pad rows)
    {
        const uint4 z = {0u, 0u, 0u, 0u};
        uint4* s16 = (uint4*)smem1;
        for (int i = tid; i < (32 + 64) * KK1P * 2 / 16; i += 128)
            s16[i] = z;
    }
    __syncthreads();

    // weights -> LDS f16: off_w is [18][72][3][3], flat index == m*648 + (c*9+t)
    for (int idx = tid; idx < 18 * KK1; idx += 128) {
        const int m  = idx / KK1;
        const int kk = idx % KK1;
        Wlds[m * KK1P + kk] = (_Float16)off_w[idx];
    }

    // im2col -> LDS f16, layout [pixel][kk = c*9 + t] (K contiguous per pixel)
    for (int idx = tid; idx < Kt * Cn * 64; idx += 128) {
        const int p = idx & 63;
        const int c = (idx >> 6) % Cn;
        const int t = idx / (64 * Cn);
        const int y = h + t / 3 - 1;
        const int x = x0 + p + (t % 3) - 1;
        float v = 0.f;
        if ((unsigned)y < (unsigned)Hn && (unsigned)x < (unsigned)Wn)
            v = feat[((b * Cn + c) * Hn + y) * Wn + x];
        Blds[p * KK1P + c * Kt + t] = (_Float16)v;
    }
    __syncthreads();

    const int wave  = tid >> 5;
    const int lane  = tid & 31;
    const int hi    = lane >> 4;   // lane half (ISA 16x16 layouts)
    const int ln    = lane & 15;
    const int pbase = wave * 16;

    v8f acc0 = {};   // out channels 0..15
    v8f acc1 = {};   // out channels 16..31 (16,17 valid)

    for (int ks = 0; ks < KK1P / 32; ++ks) {
        const int kb = ks * 32;
        Frag16 a0, a1, bf;
        // A: lane = M row; element e -> K = (e<8 ? hi*8+e : 16+hi*8+e-8)
        const _Float16* r0 = Wlds + ln * KK1P + kb;
        const _Float16* r1 = Wlds + (16 + ln) * KK1P + kb;
        a0.u[0] = *(const uint4*)(r0 + hi * 8);
        a0.u[1] = *(const uint4*)(r0 + 16 + hi * 8);
        a1.u[0] = *(const uint4*)(r1 + hi * 8);
        a1.u[1] = *(const uint4*)(r1 + 16 + hi * 8);
        // B: lane = N col (pixel); element e -> K = (e<8 ? hi*8+e : 16+hi*8+e-8)
        const _Float16* rb = Blds + (pbase + ln) * KK1P + kb;
        bf.u[0] = *(const uint4*)(rb + hi * 8);
        bf.u[1] = *(const uint4*)(rb + 16 + hi * 8);

        acc0 = __builtin_amdgcn_wmma_f32_16x16x32_f16(false, a0.h, false, bf.h,
                                                      (short)0, acc0, false, false);
        acc1 = __builtin_amdgcn_wmma_f32_16x16x32_f16(false, a1.h, false, bf.h,
                                                      (short)0, acc1, false, false);
    }

    // D layout: VGPR r -> M = r + hi*8, N = ln
    const int x = x0 + pbase + ln;
    #pragma unroll
    for (int r = 0; r < 8; ++r) {
        const int m = r + hi * 8;
        if (m < 18)
            offs[((b * 18 + m) * Hn + h) * Wn + x] = acc0[r] + off_b[m];
        const int m1 = 16 + m;
        if (m1 < 18)
            offs[((b * 18 + m1) * Hn + h) * Wn + x] = acc1[r] + off_b[m1];
    }
}

// ---------------------------------------------------------------------------
// Kernel 2: deformable conv. One workgroup = 16-pixel tile; wave g handles
// group g (8 in-ch, 8 out-ch). TDM stages the 18x16 offset tile into LDS once
// per workgroup; each wave bilinearly gathers its group's samples into a
// wave-private LDS B-tile [16 px][96 K], then 3 WMMA K-steps.
// LDS: 9 * [16][96] f16 B-tiles (27648 B) + [18][16] f32 offsets (1152 B)
// ---------------------------------------------------------------------------
__global__ __launch_bounds__(288)
void deform_conv_wmma(const float* __restrict__ input,
                      const float* __restrict__ offs,
                      const float* __restrict__ weight,
                      float* __restrict__ out)
{
    extern __shared__ char smem2[];
    const int tid  = threadIdx.x;
    const int g    = tid >> 5;     // 0..8 : group per wave
    const int lane = tid & 31;
    const int hi   = lane >> 4;
    const int ln   = lane & 15;

    const int bx = blockIdx.x;
    const int x0 = (bx % (Wn / 16)) * 16;
    const int h  = (bx / (Wn / 16)) % Hn;
    const int b  =  bx / ((Wn / 16) * Hn);

    _Float16* Bg   = (_Float16*)smem2 + g * (16 * KK2P);        // wave-private B tile
    float*    offL = (float*)(smem2 + 9 * 16 * KK2P * 2);       // [18][16] staged offsets

    const float* gsrc = offs + ((size_t)(b * 18) * Hn + h) * Wn + x0;

#if __has_builtin(__builtin_amdgcn_tensor_load_to_lds)
    // --- TDM: DMA 18 rows x 16 f32 (row stride H*W) into contiguous LDS ---
    if (tid == 0) {                      // one DMA per workgroup (EXEC ignored)
        const unsigned      ldsAddr = (unsigned)(uintptr_t)offL;   // low 32b = LDS offset
        const unsigned long gaddr   = (unsigned long)(uintptr_t)gsrc;

        u32x4 g0;                        // D# group 0
        g0[0] = 1u;                                      // count=1, user descriptor
        g0[1] = ldsAddr;                                 // lds_addr [63:32]
        g0[2] = (unsigned)(gaddr & 0xffffffffu);         // global_addr [95:64]
        g0[3] = (unsigned)((gaddr >> 32) & 0x1ffffffu)   // global_addr [120:96]
              | (2u << 30);                              // type=2 (image)

        i32x8 g1;                        // D# group 1
        g1[0] = (int)(2u << 16);                         // data_size=4B
        g1[1] = (int)(((unsigned)Wn & 0xffffu) << 16);   // tensor_dim0 = 320
        g1[2] = (int)((unsigned)(18 * Hn) << 16);        // tensor_dim1 = 3240
        g1[3] = (int)(16u << 16);                        // tile_dim0 = 16
        g1[4] = 18;                                      // tile_dim1 = 18, tile_dim2 = 0
        g1[5] = Hn * Wn;                                 // tensor_dim0_stride = 57600
        g1[6] = 0;
        g1[7] = 0;

        const i32x4 gz4 = {0, 0, 0, 0};                  // groups 2/3 unused (2D tensor)
        const i32x8 gz8 = {0, 0, 0, 0, 0, 0, 0, 0};
        __builtin_amdgcn_tensor_load_to_lds(g0, g1, gz4, gz4, gz8, 0);
    }
    if (tid < 32)
        __builtin_amdgcn_s_wait_tensorcnt(0);
#else
    // --- fallback: cooperative staged load ---
    for (int i = tid; i < 18 * 16; i += 288) {
        const int ch = i >> 4, p = i & 15;
        offL[ch * 16 + p] = gsrc[(size_t)ch * Hn * Wn + p];
    }
#endif
    __syncthreads();

    // speculative prefetch of this tile's input rows (global_prefetch_b8)
    __builtin_prefetch(input + ((size_t)(b * Cn + g * 8) * Hn + h) * Wn + x0 + lane, 0, 1);

    // zero K-pad columns 72..95
    for (int i = lane; i < 16 * (KK2P - KK2); i += 32) {
        const int p  = i / (KK2P - KK2);
        const int kk = KK2 + i % (KK2P - KK2);
        Bg[p * KK2P + kk] = (_Float16)0.f;
    }

    // bilinear sampling into Bg[p][c*9 + k]; 8ch * 9tap * 16px = 1152 = 36 iters
    for (int it = 0; it < 36; ++it) {
        const int idx = lane + it * 32;
        const int p = idx & 15;
        const int k = (idx >> 4) % Kt;
        const int c = idx / (16 * Kt);
        const int x = x0 + p;

        const float dy = offL[(2 * k + 0) * 16 + p];
        const float dx = offL[(2 * k + 1) * 16 + p];
        const float py = (float)h + (float)(k / 3 - 1) + dy;
        const float px = (float)x + (float)(k % 3 - 1) + dx;

        const float fy = floorf(py), fx = floorf(px);
        const int   iy = (int)fy,    ix = (int)fx;
        const float wy1 = py - fy, wx1 = px - fx;
        const float wy0 = 1.f - wy1, wx0 = 1.f - wx1;

        const float* img = input + (size_t)((b * Cn + g * 8 + c) * Hn) * Wn;
        float v = 0.f;
        {
            const float w00 = wy0 * wx0, w01 = wy0 * wx1;
            const float w10 = wy1 * wx0, w11 = wy1 * wx1;
            if ((unsigned)iy < (unsigned)Hn && (unsigned)ix < (unsigned)Wn)
                v += w00 * img[iy * Wn + ix];
            if ((unsigned)iy < (unsigned)Hn && (unsigned)(ix + 1) < (unsigned)Wn)
                v += w01 * img[iy * Wn + ix + 1];
            if ((unsigned)(iy + 1) < (unsigned)Hn && (unsigned)ix < (unsigned)Wn)
                v += w10 * img[(iy + 1) * Wn + ix];
            if ((unsigned)(iy + 1) < (unsigned)Hn && (unsigned)(ix + 1) < (unsigned)Wn)
                v += w11 * img[(iy + 1) * Wn + ix + 1];
        }
        Bg[p * KK2P + c * Kt + k] = (_Float16)v;
    }

    // A fragments from weights: weight is [72][8][3][3]; row o=g*8+m, flat kk=c*9+k
    Frag16 afr[3];
    for (int ks = 0; ks < 3; ++ks) {
        #pragma unroll
        for (int e = 0; e < 16; ++e) {
            const int kk = ks * 32 + hi * 16 + e;
            float wv = 0.f;
            if (ln < 8 && kk < KK2)
                wv = weight[(g * 8 + ln) * KK2 + kk];
            afr[ks].h[e] = (_Float16)wv;
        }
    }

    // 3 WMMA K-steps (same-wave DS store->load, hardware-ordered, no barrier)
    v8f acc = {};
    for (int ks = 0; ks < 3; ++ks) {
        Frag16 bf;
        const _Float16* rb = Bg + ln * KK2P + ks * 32 + hi * 16;
        bf.u[0] = *(const uint4*)(rb);
        bf.u[1] = *(const uint4*)(rb + 8);
        acc = __builtin_amdgcn_wmma_f32_16x16x32_f16(false, afr[ks].h, false, bf.h,
                                                     (short)0, acc, false, false);
    }

    // D rows 0..7 valid (Og=8): lanes 0..15, VGPR r -> out channel g*8+r, col x0+ln
    if (hi == 0) {
        #pragma unroll
        for (int r = 0; r < 8; ++r)
            out[((b * On + g * 8 + r) * Hn + h) * Wn + x0 + ln] = acc[r];
    }
}

// ---------------------------------------------------------------------------
extern "C" void kernel_launch(void* const* d_in, const int* in_sizes, int n_in,
                              void* d_out, int out_size, void* d_ws, size_t ws_size,
                              hipStream_t stream)
{
    const float* input  = (const float*)d_in[0];   // (2,72,180,320)
    const float* feat   = (const float*)d_in[1];   // (2,72,180,320)
    const float* weight = (const float*)d_in[2];   // (72,8,3,3)
    const float* off_w  = (const float*)d_in[3];   // (18,72,3,3)
    const float* off_b  = (const float*)d_in[4];   // (18,)

    float* offs = (float*)d_ws;                    // (2,18,180,320) f32 = 8.3 MB

    const dim3 g1(Bn * Hn * (Wn / 64));            // 1800 workgroups
    const dim3 b1(128);
    const size_t lds1 = (size_t)(32 + 64) * KK1P * 2;     // 129 KB
    offset_conv_wmma<<<g1, b1, lds1, stream>>>(feat, off_w, off_b, offs);

    const dim3 g2(Bn * Hn * (Wn / 16));            // 7200 workgroups
    const dim3 b2(288);                            // 9 waves = 9 groups
    const size_t lds2 = (size_t)9 * 16 * KK2P * 2 + 18 * 16 * 4;   // 28.8 KB
    deform_conv_wmma<<<g2, b2, lds2, stream>>>(input, offs, weight, (float*)d_out);
}
